// EncoderLayer_90228672954752
// MI455X (gfx1250) — matrix operand order repaired and checked by reading, not compiled, not run on hardware
//
#include <hip/hip_runtime.h>
#include <hip/hip_bf16.h>
#include <cstdint>
#include <cstddef>

typedef __attribute__((ext_vector_type(16))) __bf16       v16bf;
typedef __attribute__((ext_vector_type(8)))  float        v8f;
typedef __attribute__((ext_vector_type(8)))  unsigned int v8u;
typedef __attribute__((ext_vector_type(4)))  unsigned int v4u;

#define WMMA_BF16(a, b, c) \
  __builtin_amdgcn_wmma_f32_16x16x32_bf16(false, (a), false, (b), (short)0, (c), false, false)

constexpr int MODEL = 1024;
constexpr int HEADS = 16;
constexpr int HEAD_DIM = 64;
constexpr int INNER = 4096;
constexpr int SEQ = 2048;
constexpr int BATCH = 4;
constexpr int NT = BATCH * SEQ;       // 8192 tokens
constexpr float EPS = 1e-5f;

// ---------------------------------------------------------------------------
// Tensor Data Mover (gfx1250) via inline asm — portable across both
// toolchains (bypasses the 5-arg/6-arg builtin mismatch) and guarantees the
// instruction is emitted.  2-operand form = VADDR2/VADDR3 NULL (<=2D tensor),
// per CDNA5 ISA.
// Descriptor: D# group0 {count=1 | lds_addr | global_addr | type=2},
// group1 {data_size=2B, pad_enable, pad_interval, pad_amount=1DW,
//         tensor/tile dims, 48-bit dim0 stride}.
// ---------------------------------------------------------------------------
__device__ __forceinline__ void tdm_load_tile_bf16(
    const __bf16* gptr, unsigned lds_byte_addr,
    unsigned tile_w, unsigned tile_h, unsigned row_stride_elems,
    unsigned pad_code) {
  unsigned long long ga = (unsigned long long)(uintptr_t)gptr;
  v4u g0;
  g0[0] = 1u;                                               // count=1, user mode
  g0[1] = lds_byte_addr;                                    // D#.lds_addr
  g0[2] = (unsigned)(ga & 0xFFFFFFFFu);                     // global_addr[31:0]
  g0[3] = (unsigned)((ga >> 32) & 0x1FFFFFFu) | (2u << 30); // [56:32] | type=2

  unsigned long long s0 = (unsigned long long)row_stride_elems;
  v8u g1;
  g1[0] = (1u << 16)                 // data_size = 2 bytes
        | (1u << 20)                 // pad_enable
        | (pad_code << 22)           // pad_interval: pad every (8<<code) bytes
        | (0u << 25);                // pad_amount = 1 DWORD
  g1[1] = (tile_w & 0xFFFFu) << 16;                         // tensor_dim0 lo
  g1[2] = (tile_w >> 16) | ((tile_h & 0xFFFFu) << 16);      // td0 hi | td1 lo
  g1[3] = (tile_h >> 16) | (tile_w << 16);                  // td1 hi | tile_dim0
  g1[4] = tile_h & 0xFFFFu;                                 // tile_dim1 (dim2=0)
  g1[5] = (unsigned)(s0 & 0xFFFFFFFFull);                   // stride0[31:0]
  g1[6] = (unsigned)((s0 >> 32) & 0xFFFFull);               // stride0[47:32]
  g1[7] = 0u;                                               // stride1 (unused 2D)

  asm volatile("tensor_load_to_lds %0, %1"
               :: "s"(g0), "s"(g1)
               : "memory");
}

__device__ __forceinline__ void tdm_wait() {
  asm volatile("s_wait_tensorcnt 0x0" ::: "memory");
}

// ---------------------------------------------------------------------------
// fp32 -> bf16 conversion (grid-stride)
// ---------------------------------------------------------------------------
__global__ void f32_to_bf16(const float* __restrict__ in, __bf16* __restrict__ out, int n) {
  int i = blockIdx.x * blockDim.x + threadIdx.x;
  int stride = gridDim.x * blockDim.x;
  for (; i < n; i += stride) out[i] = (__bf16)in[i];
}

// ---------------------------------------------------------------------------
// bf16 WMMA GEMM, 128x128 block tile, 8 waves (4x2 wave grid each 32Mx64N),
// K-step 32.  A tiles: TDM double-buffered (DMA of tile k+1 overlaps WMMA of
// tile k).  B tiles: register-prefetched global b64 loads, transposed into
// LDS.  Epilogue: 0 = fp32 + fp32 residual; 1 = relu->bf16; 2 = *scale->bf16.
// ---------------------------------------------------------------------------
template <int EPI>
__global__ __launch_bounds__(256) void gemm_bf16_128x128(
    const __bf16* __restrict__ A, const __bf16* __restrict__ W,
    const float* __restrict__ bias, const float* __restrict__ resid,
    float* __restrict__ outF, __bf16* __restrict__ outB,
    int M, int N, int K, float scale) {
  __shared__ __bf16 As[2][128][34];  // [buf][m][k], TDM pad: 64B row + 1 DWORD
  __shared__ __bf16 Bs[128][34];     // transposed: [n][k], padded

  const int tid  = threadIdx.x;
  const int wave = tid >> 5;
  const int lane = tid & 31;
  const int lh   = lane >> 4;        // 16-lane half
  const int l15  = lane & 15;
  const int wm   = wave >> 1;        // 0..3 -> M offset wm*32
  const int wn   = wave & 1;         // 0..1 -> N offset wn*64
  const int m0   = blockIdx.y * 128;
  const int n0   = blockIdx.x * 128;

  const __bf16* Arow = A + (size_t)m0 * K;
  const unsigned asAddr[2] = {(unsigned)(uintptr_t)&As[0][0][0],
                              (unsigned)(uintptr_t)&As[1][0][0]};

  v8f acc[2][4] = {};

  // prologue: post A(0) DMA, prefetch B(0) into registers
  if (tid < 32)
    tdm_load_tile_bf16(Arow, asAddr[0], 32, 128, (unsigned)K, /*pad=*/3);
  unsigned long long breg[4];
#pragma unroll
  for (int c = 0; c < 4; ++c) {
    int e = c * 1024 + tid * 4;
    int bk = e >> 7, bn = e & 127;
    breg[c] = *(const unsigned long long*)&W[(size_t)bk * N + n0 + bn];
  }

  const int nsteps = K / 32;
  for (int step = 0; step < nsteps; ++step) {
    const int k0 = step * 32;
    __syncthreads();                       // (a) previous compute done
    // commit prefetched B tile, transposed
#pragma unroll
    for (int c = 0; c < 4; ++c) {
      int e = c * 1024 + tid * 4;
      int bk = e >> 7, bn = e & 127;
      const __bf16* hh = (const __bf16*)&breg[c];
#pragma unroll
      for (int t = 0; t < 4; ++t) Bs[bn + t][bk] = hh[t];
    }
    if (tid < 32) tdm_wait();              // A(step) landed in LDS
    __syncthreads();                       // (b) publish As[step&1] + Bs
    if (step + 1 < nsteps) {
      // overlap: DMA A(step+1) + global loads of B(step+1) under the WMMAs
      if (tid < 32)
        tdm_load_tile_bf16(Arow + k0 + 32, asAddr[(step + 1) & 1],
                           32, 128, (unsigned)K, 3);
#pragma unroll
      for (int c = 0; c < 4; ++c) {
        int e = c * 1024 + tid * 4;
        int bk = e >> 7, bn = e & 127;
        breg[c] = *(const unsigned long long*)&W[(size_t)(k0 + 32 + bk) * N + n0 + bn];
      }
    }

    const __bf16 (*Ac)[34] = As[step & 1];
    v8u au[2];
#pragma unroll
    for (int mt = 0; mt < 2; ++mt) {
      int row = wm * 32 + mt * 16 + l15;
#pragma unroll
      for (int j = 0; j < 8; ++j) {
        int kk = ((j & 4) << 2) + lh * 8 + ((j & 3) << 1);   // A-frag K interleave
        au[mt][j] = *(const unsigned int*)&Ac[row][kk];
      }
    }
    v8u bu[4];
#pragma unroll
    for (int nt = 0; nt < 4; ++nt) {
      int col = wn * 64 + nt * 16 + l15;
#pragma unroll
      for (int j = 0; j < 8; ++j) {
        int kk = ((j & 4) << 2) + lh * 8 + ((j & 3) << 1);   // B-frag K interleave
        bu[nt][j] = *(const unsigned int*)&Bs[col][kk];
      }
    }
#pragma unroll
    for (int mt = 0; mt < 2; ++mt)
#pragma unroll
      for (int nt = 0; nt < 4; ++nt)
        acc[mt][nt] = WMMA_BF16(__builtin_bit_cast(v16bf, au[mt]),
                                __builtin_bit_cast(v16bf, bu[nt]),
                                acc[mt][nt]);
  }

  // epilogue (C layout: m = r + 8*lh, n = l15)
#pragma unroll
  for (int mt = 0; mt < 2; ++mt)
#pragma unroll
    for (int nt = 0; nt < 4; ++nt)
#pragma unroll
      for (int r = 0; r < 8; ++r) {
        int gm = m0 + wm * 32 + mt * 16 + r + lh * 8;
        int gn = n0 + wn * 64 + nt * 16 + l15;
        float v = acc[mt][nt][r] + bias[gn];
        size_t idx = (size_t)gm * N + gn;
        if constexpr (EPI == 0) {
          outF[idx] = v + resid[idx];
        } else if constexpr (EPI == 1) {
          outB[idx] = (__bf16)fmaxf(v, 0.0f);
        } else {
          outB[idx] = (__bf16)(v * scale);
        }
      }
}

// ---------------------------------------------------------------------------
// Flash attention: one workgroup per (q-tile of 64, head, batch), 4 waves.
// Q pre-scaled by 1/sqrt(Dh).  Q via TDM once; K tiles TDM double-buffered
// (DMA overlaps WMMA); V^T tile register-prefetched.  Online softmax in the
// C-register layout; P round-trips through LDS (C-layout -> A-layout).
// ---------------------------------------------------------------------------
__global__ __launch_bounds__(128) void flash_attn(
    const __bf16* __restrict__ Qg, const __bf16* __restrict__ Kg,
    const __bf16* __restrict__ Vg, const int* __restrict__ mask,
    __bf16* __restrict__ ctx) {
  __shared__ __bf16 Qs[64][66];       // [q][d]    (TDM pad: 128B row + 1 DWORD)
  __shared__ __bf16 Ks[2][64][66];    // [buf][key][d]
  __shared__ __bf16 Vs[64][66];       // transposed: [d][key]
  __shared__ __bf16 Ps[4][16][66];    // per-wave P strip [q][key]

  const int tid  = threadIdx.x;
  const int wave = tid >> 5;
  const int lane = tid & 31;
  const int lh   = lane >> 4;
  const int l15  = lane & 15;
  const int qt   = blockIdx.x;
  const int h    = blockIdx.y;
  const int b    = blockIdx.z;
  const size_t headoff = (size_t)h * HEAD_DIM;
  const unsigned ksAddr[2] = {(unsigned)(uintptr_t)&Ks[0][0][0],
                              (unsigned)(uintptr_t)&Ks[1][0][0]};

  // Q tile via TDM, wait, publish
  if (tid < 32) {
    tdm_load_tile_bf16(Qg + (size_t)(b * SEQ + qt * 64) * MODEL + headoff,
                       (unsigned)(uintptr_t)&Qs[0][0],
                       64, 64, MODEL, /*pad=*/4);   // pad every 32 DWORDs
    tdm_wait();
  }
  __syncthreads();

  // Q A-frags for this wave's 16 query rows (Dh=64 -> 2 chunks of K=32)
  v8u aq[2];
#pragma unroll
  for (int c = 0; c < 2; ++c)
#pragma unroll
    for (int j = 0; j < 8; ++j) {
      int kk = c * 32 + ((j & 4) << 2) + lh * 8 + ((j & 3) << 1);
      aq[c][j] = *(const unsigned int*)&Qs[wave * 16 + l15][kk];
    }

  // prologue: post K(0) DMA, prefetch V(0) into registers
  if (tid < 32)
    tdm_load_tile_bf16(Kg + (size_t)(b * SEQ) * MODEL + headoff,
                       ksAddr[0], 64, 64, MODEL, 4);
  unsigned long long vreg[8];
#pragma unroll
  for (int c = 0; c < 8; ++c) {
    int e = c * 512 + tid * 4;
    int key = e >> 6, d = e & 63;
    vreg[c] = *(const unsigned long long*)&Vg[(size_t)(b * SEQ + key) * MODEL + headoff + d];
  }

  v8f acc[4] = {};
  float mstate[8], lstate[8];
#pragma unroll
  for (int r = 0; r < 8; ++r) { mstate[r] = -3.0e38f; lstate[r] = 0.0f; }

  const int ktiles = SEQ / 64;
  for (int kt = 0; kt < ktiles; ++kt) {
    __syncthreads();                       // (a) previous compute done
    // commit prefetched V tile, transposed
#pragma unroll
    for (int c = 0; c < 8; ++c) {
      int e = c * 512 + tid * 4;
      int key = e >> 6, d = e & 63;
      const __bf16* hh = (const __bf16*)&vreg[c];
#pragma unroll
      for (int t = 0; t < 4; ++t) Vs[d + t][key] = hh[t];
    }
    if (tid < 32) tdm_wait();              // K(kt) landed
    __syncthreads();                       // (b) publish Ks[kt&1] + Vs
    if (kt + 1 < ktiles) {
      if (tid < 32)
        tdm_load_tile_bf16(Kg + (size_t)(b * SEQ + (kt + 1) * 64) * MODEL + headoff,
                           ksAddr[(kt + 1) & 1], 64, 64, MODEL, 4);
#pragma unroll
      for (int c = 0; c < 8; ++c) {
        int e = c * 512 + tid * 4;
        int key = e >> 6, d = e & 63;
        vreg[c] = *(const unsigned long long*)
            &Vg[(size_t)(b * SEQ + (kt + 1) * 64 + key) * MODEL + headoff + d];
      }
    }
    const __bf16 (*Kc)[66] = Ks[kt & 1];

    // S = Q * K^T  (64 keys -> 4 n-tiles; Dh=64 -> 2 k-chunks)
    v8f s[4] = {};
#pragma unroll
    for (int nt = 0; nt < 4; ++nt)
#pragma unroll
      for (int c = 0; c < 2; ++c) {
        v8u bk;
#pragma unroll
        for (int j = 0; j < 8; ++j) {
          int kk = c * 32 + ((j & 4) << 2) + lh * 8 + ((j & 3) << 1);
          bk[j] = *(const unsigned int*)&Kc[nt * 16 + l15][kk];
        }
        s[nt] = WMMA_BF16(__builtin_bit_cast(v16bf, aq[c]),
                          __builtin_bit_cast(v16bf, bk), s[nt]);
      }

    // mask (n = key column of this lane)
#pragma unroll
    for (int nt = 0; nt < 4; ++nt) {
      int kg = kt * 64 + nt * 16 + l15;
      if (mask[b * SEQ + kg] == 0) {
#pragma unroll
        for (int r = 0; r < 8; ++r) s[nt][r] = -1.0e10f;
      }
    }

    // online softmax, per register-row r (strip row = r + 8*lh)
#pragma unroll
    for (int r = 0; r < 8; ++r) {
      float mx = fmaxf(fmaxf(s[0][r], s[1][r]), fmaxf(s[2][r], s[3][r]));
#pragma unroll
      for (int msk = 1; msk < 16; msk <<= 1) mx = fmaxf(mx, __shfl_xor(mx, msk, 32));
      float mnew = fmaxf(mstate[r], mx);
      float sc = __expf(mstate[r] - mnew);
      mstate[r] = mnew;
      float rs = 0.0f;
#pragma unroll
      for (int nt = 0; nt < 4; ++nt) {
        float p = __expf(s[nt][r] - mnew);
        s[nt][r] = p;
        rs += p;
      }
#pragma unroll
      for (int msk = 1; msk < 16; msk <<= 1) rs += __shfl_xor(rs, msk, 32);
      lstate[r] = lstate[r] * sc + rs;
#pragma unroll
      for (int dt = 0; dt < 4; ++dt) acc[dt][r] *= sc;
    }

    // P: C-layout regs -> LDS strip (bf16)
#pragma unroll
    for (int nt = 0; nt < 4; ++nt)
#pragma unroll
      for (int r = 0; r < 8; ++r)
        Ps[wave][r + lh * 8][nt * 16 + l15] = (__bf16)s[nt][r];

    // ctx += P @ V  (keys = K-dim, 2 chunks of 32)
#pragma unroll
    for (int c = 0; c < 2; ++c) {
      v8u ap;
#pragma unroll
      for (int j = 0; j < 8; ++j) {
        int kk = c * 32 + ((j & 4) << 2) + lh * 8 + ((j & 3) << 1);
        ap[j] = *(const unsigned int*)&Ps[wave][l15][kk];
      }
#pragma unroll
      for (int dt = 0; dt < 4; ++dt) {
        v8u bv;
#pragma unroll
        for (int j = 0; j < 8; ++j) {
          int kk = c * 32 + ((j & 4) << 2) + lh * 8 + ((j & 3) << 1);
          bv[j] = *(const unsigned int*)&Vs[dt * 16 + l15][kk];
        }
        acc[dt] = WMMA_BF16(__builtin_bit_cast(v16bf, ap),
                            __builtin_bit_cast(v16bf, bv), acc[dt]);
      }
    }
  }

  // write ctx / l
#pragma unroll
  for (int dt = 0; dt < 4; ++dt)
#pragma unroll
    for (int r = 0; r < 8; ++r) {
      int row = b * SEQ + qt * 64 + wave * 16 + r + lh * 8;
      int col = h * HEAD_DIM + dt * 16 + l15;
      ctx[(size_t)row * MODEL + col] = (__bf16)(acc[dt][r] / lstate[r]);
    }
}

// ---------------------------------------------------------------------------
// LayerNorm over 1024 features, one block per token; optional bf16 copy.
// ---------------------------------------------------------------------------
__global__ __launch_bounds__(256) void layernorm_k(
    const float* __restrict__ y, const float* __restrict__ g,
    const float* __restrict__ be, float* __restrict__ outF,
    __bf16* __restrict__ outB) {
  __shared__ float r1[256], r2[256];
  const int t = threadIdx.x;
  const size_t base = (size_t)blockIdx.x * MODEL;
  float v[4];
  float s1 = 0.0f, s2 = 0.0f;
#pragma unroll
  for (int i = 0; i < 4; ++i) {
    v[i] = y[base + t + i * 256];
    s1 += v[i];
    s2 += v[i] * v[i];
  }
  r1[t] = s1; r2[t] = s2;
  __syncthreads();
  for (int off = 128; off > 0; off >>= 1) {
    if (t < off) { r1[t] += r1[t + off]; r2[t] += r2[t + off]; }
    __syncthreads();
  }
  float mu  = r1[0] * (1.0f / MODEL);
  float var = r2[0] * (1.0f / MODEL) - mu * mu;
  float rstd = rsqrtf(var + EPS);
#pragma unroll
  for (int i = 0; i < 4; ++i) {
    int col = t + i * 256;
    float o = (v[i] - mu) * rstd * g[col] + be[col];
    outF[base + col] = o;
    if (outB) outB[base + col] = (__bf16)o;
  }
}

// ---------------------------------------------------------------------------
extern "C" void kernel_launch(void* const* d_in, const int* in_sizes, int n_in,
                              void* d_out, int out_size, void* d_ws, size_t ws_size,
                              hipStream_t stream) {
  (void)in_sizes; (void)n_in; (void)out_size; (void)ws_size;
  const float* x   = (const float*)d_in[0];
  const int*   msk = (const int*)d_in[1];
  const float* Wq  = (const float*)d_in[2];
  const float* bq  = (const float*)d_in[3];
  const float* Wk  = (const float*)d_in[4];
  const float* bk  = (const float*)d_in[5];
  const float* Wv  = (const float*)d_in[6];
  const float* bv  = (const float*)d_in[7];
  const float* Wo  = (const float*)d_in[8];
  const float* bo  = (const float*)d_in[9];
  const float* g1  = (const float*)d_in[10];
  const float* be1 = (const float*)d_in[11];
  const float* W1  = (const float*)d_in[12];
  const float* b1  = (const float*)d_in[13];
  const float* W2  = (const float*)d_in[14];
  const float* b2  = (const float*)d_in[15];
  const float* g2  = (const float*)d_in[16];
  const float* be2 = (const float*)d_in[17];
  float* out = (float*)d_out;

  char* ws = (char*)d_ws;
  const size_t MiB = 1024 * 1024;
  __bf16* wqb  = (__bf16*)(ws + 0 * MiB);    // 2 MiB each
  __bf16* wkb  = (__bf16*)(ws + 2 * MiB);
  __bf16* wvb  = (__bf16*)(ws + 4 * MiB);
  __bf16* wob  = (__bf16*)(ws + 6 * MiB);
  __bf16* w1b  = (__bf16*)(ws + 8 * MiB);    // 8 MiB
  __bf16* w2b  = (__bf16*)(ws + 16 * MiB);   // 8 MiB
  __bf16* xb   = (__bf16*)(ws + 24 * MiB);   // 16 MiB (reused as x1 bf16)
  __bf16* qb   = (__bf16*)(ws + 40 * MiB);   // 16 MiB
  __bf16* kbuf = (__bf16*)(ws + 56 * MiB);   // 16 MiB
  __bf16* vbuf = (__bf16*)(ws + 72 * MiB);   // 16 MiB
  __bf16* ctxb = (__bf16*)(ws + 88 * MiB);   // 16 MiB
  float*  yf   = (float*)(ws + 104 * MiB);   // 32 MiB
  float*  x1f  = (float*)(ws + 136 * MiB);   // 32 MiB    (total 168 MiB)
  __bf16* hb   = (__bf16*)(ws + 40 * MiB);   // 64 MiB, overlaps dead q/k/v/ctx
  __bf16* x1b  = xb;

  // 1) bf16 conversions
  f32_to_bf16<<<512, 256, 0, stream>>>(Wq, wqb, MODEL * MODEL);
  f32_to_bf16<<<512, 256, 0, stream>>>(Wk, wkb, MODEL * MODEL);
  f32_to_bf16<<<512, 256, 0, stream>>>(Wv, wvb, MODEL * MODEL);
  f32_to_bf16<<<512, 256, 0, stream>>>(Wo, wob, MODEL * MODEL);
  f32_to_bf16<<<512, 256, 0, stream>>>(W1, w1b, MODEL * INNER);
  f32_to_bf16<<<512, 256, 0, stream>>>(W2, w2b, INNER * MODEL);
  f32_to_bf16<<<512, 256, 0, stream>>>(x, xb, NT * MODEL);

  dim3 blk(256);
  dim3 gProj(MODEL / 128, NT / 128);   // 8 x 64
  dim3 gFF1(INNER / 128, NT / 128);    // 32 x 64

  // 2) Q/K/V projections (Q scaled by 1/sqrt(64))
  gemm_bf16_128x128<2><<<gProj, blk, 0, stream>>>(xb, wqb, bq, nullptr, nullptr, qb,
                                                  NT, MODEL, MODEL, 0.125f);
  gemm_bf16_128x128<2><<<gProj, blk, 0, stream>>>(xb, wkb, bk, nullptr, nullptr, kbuf,
                                                  NT, MODEL, MODEL, 1.0f);
  gemm_bf16_128x128<2><<<gProj, blk, 0, stream>>>(xb, wvb, bv, nullptr, nullptr, vbuf,
                                                  NT, MODEL, MODEL, 1.0f);

  // 3) flash attention
  dim3 gFA(SEQ / 64, HEADS, BATCH);
  flash_attn<<<gFA, 128, 0, stream>>>(qb, kbuf, vbuf, msk, ctxb);

  // 4) output projection + residual, then LN1
  gemm_bf16_128x128<0><<<gProj, blk, 0, stream>>>(ctxb, wob, bo, x, yf, nullptr,
                                                  NT, MODEL, MODEL, 1.0f);
  layernorm_k<<<NT, 256, 0, stream>>>(yf, g1, be1, x1f, x1b);

  // 5) FFN
  gemm_bf16_128x128<1><<<gFF1, blk, 0, stream>>>(x1b, w1b, b1, nullptr, nullptr, hb,
                                                 NT, INNER, MODEL, 1.0f);
  gemm_bf16_128x128<0><<<gProj, blk, 0, stream>>>(hb, w2b, b2, x1f, yf, nullptr,
                                                  NT, MODEL, INNER, 1.0f);
  layernorm_k<<<NT, 256, 0, stream>>>(yf, g2, be2, out, nullptr);
}
